// row_attention_perlevel_0_91139206021613
// MI455X (gfx1250) — compile-verified, matrix operand order checked by reading, not run on hardware
//
#include <hip/hip_runtime.h>
#include <hip/hip_bf16.h>
#include <cstdint>

typedef __attribute__((ext_vector_type(16))) _Float16 v16h;
typedef __attribute__((ext_vector_type(8)))  float    v8f;

#define BB     2
#define CC     256
#define REPC   128
#define HH     64
#define WWID   256
#define NBLOCKROWS 32
#define HWIMG  (HH*WWID)        // 16384
#define NQ     768
#define NQOUT  512
#define NBLK   128              // 2 sides * 2 batch * 32 blocks

__device__ __forceinline__ v16h ld16h(const _Float16* p){ return *(const v16h*)p; }

__device__ __forceinline__ v8f wmma16(v16h a, v16h b, v8f c){
  // D = A(16x32 f16) * B(32x16 f16) + C(16x16 f32)
  return __builtin_amdgcn_wmma_f32_16x16x32_f16(false, a, false, b, (short)0, c, false, false);
}

// async global -> LDS copy, 16 bytes per lane (tracked by ASYNCcnt)
__device__ __forceinline__ void async_g2l_b128(const void* gptr, unsigned lds_off){
  asm volatile("global_load_async_to_lds_b128 %0, %1, off"
               :: "v"(lds_off), "v"((unsigned long long)(uintptr_t)gptr)
               : "memory");
}
__device__ __forceinline__ void wait_async0(){
  asm volatile("s_wait_asynccnt 0" ::: "memory");
}
__device__ __forceinline__ unsigned lds_off_of(const void* p){
  return (unsigned)(uintptr_t)p;   // low 32 bits of LDS-aperture address = LDS offset
}

// map token q (0..767) of block i to flattened (h,w) of the source image,
// honoring the reflect pad (padded row 64 -> source row 62)
__device__ __forceinline__ int qhw(int i, int q){
  int r  = q >> 8;
  int pr = 2*i + r;
  int sr = (pr > 63) ? 62 : pr;
  return sr*WWID + (q & 255);
}

// ---------------------------------------------------------------------------
// Kernel 0: fp16 weight copies + BatchNorm folding
// ---------------------------------------------------------------------------
__global__ void prep_kernel(const float* Wt, const float* Wp, const float* Wg,
                            const float* Wu, const float* bu, const float* gamma,
                            const float* beta, const float* rmean, const float* rvar,
                            _Float16* wth, _Float16* wph, _Float16* wgh, _Float16* wuh,
                            float* effs, float* effb)
{
  const int NW = REPC*CC;            // 32768
  int t = blockIdx.x*blockDim.x + threadIdx.x;
  if      (t <   NW) wth[t      ] = (_Float16)Wt[t];
  else if (t < 2*NW) wph[t -  NW] = (_Float16)Wp[t -   NW];
  else if (t < 3*NW) wgh[t -2*NW] = (_Float16)Wg[t - 2*NW];
  else if (t < 4*NW) wuh[t -3*NW] = (_Float16)Wu[t - 3*NW];
  if (t < CC){
    float inv = gamma[t] * rsqrtf(rvar[t] + 1e-5f);
    effs[t] = inv;
    effb[t] = beta[t] + (bu[t] - rmean[t]) * inv;
  }
}

// ---------------------------------------------------------------------------
// Kernel 1: projections  proj[rep, hw] = W[rep,c] @ x[c,hw] + bias
//   proj 0 (T) and 1 (P): stored transposed  [hw][rep]  (A / B-of-S operands)
//   proj 2 (G):           stored natural     [rep][hw]  (B-of-O operand)
// grid = img(4) * proj(3) * hw-tiles(128); 256 threads = 8 waves
// ---------------------------------------------------------------------------
__global__ void proj_kernel(const float* left, const float* right,
                            const _Float16* wth, const _Float16* wph, const _Float16* wgh,
                            const float* bt, const float* bp, const float* bg,
                            _Float16* Tp, _Float16* Pp, _Float16* Gp)
{
  __shared__ _Float16 ldsX[128*32];          // [hw_local][c_local]

  const int tile = blockIdx.x & 127;
  const int proj = (blockIdx.x >> 7) % 3;
  const int img  = blockIdx.x / 384;         // img = side*2 + batch
  const int side = img >> 1, batch = img & 1;

  const float* src = (side ? right : left) + (size_t)batch*CC*HWIMG;
  const _Float16* wb = (proj == 0) ? wth : (proj == 1) ? wph : wgh;
  const float*    bb = (proj == 0) ? bt  : (proj == 1) ? bp  : bg;

  const int n0   = tile*128;
  const int tid  = threadIdx.x;
  const int lane = tid & 31, wave = tid >> 5;
  const int l16  = lane & 15, hg = lane >> 4;

  v8f acc[8];
  const v8f vz = {0.f,0.f,0.f,0.f,0.f,0.f,0.f,0.f};
#pragma unroll
  for (int n = 0; n < 8; ++n) acc[n] = vz;

  for (int kc = 0; kc < 8; ++kc){            // c in chunks of 32
    // stage x chunk transposed into LDS: [hw 128][c 32]
    int hwl = tid >> 1, cp = (tid & 1)*16;
#pragma unroll
    for (int j = 0; j < 16; ++j){
      int c = kc*32 + cp + j;
      ldsX[hwl*32 + cp + j] = (_Float16)src[(size_t)c*HWIMG + n0 + hwl];
    }
    __syncthreads();

    v16h a = ld16h(&wb[(size_t)(wave*16 + l16)*CC + kc*32 + hg*16]);
#pragma unroll
    for (int n = 0; n < 8; ++n){
      v16h b = ld16h(&ldsX[(n*16 + l16)*32 + hg*16]);
      acc[n] = wmma16(a, b, acc[n]);
    }
    __syncthreads();
  }

  // epilogue: bias + store
  _Float16* outT = (proj == 0 ? Tp : Pp) + (size_t)img*HWIMG*REPC;  // [hw][rep]
  _Float16* outG = Gp + (size_t)img*REPC*HWIMG;                      // [rep][hw]
#pragma unroll
  for (int e = 0; e < 8; ++e){
    int repr = wave*16 + e + 8*hg;
    float bv = bb[repr];
#pragma unroll
    for (int n = 0; n < 8; ++n){
      int col = n0 + n*16 + l16;
      float v = acc[n][e] + bv;
      if (proj < 2) outT[(size_t)col*REPC + repr]  = (_Float16)v;
      else          outG[(size_t)repr*HWIMG + col] = (_Float16)v;
    }
  }
}

// ---------------------------------------------------------------------------
// Kernel 2: per-block attention (online softmax without max subtraction).
// grid = NBLK(128) * qtiles(2); 256 threads = 8 waves; wave owns 32 queries.
//   S = TH(q,d) @ PH(d,k), P = exp(S), O = P @ G, maxi = (P·col)/Σ
// PH/G key chunks staged with async global->LDS (ASYNCcnt) copies.
// ---------------------------------------------------------------------------
__global__ void attn_kernel(const _Float16* Tp, const _Float16* Pp, const _Float16* Gp,
                            _Float16* after, float* maxi_ws)
{
  __shared__ _Float16 ldsP[32*128];           // [key][d]
  __shared__ _Float16 ldsG[128*32];           // [d][key]
  __shared__ _Float16 ldsProb[8][32*32];      // per-wave [q_local][key_local]

  const int qtile = blockIdx.x & 1;
  const int blk   = blockIdx.x >> 1;
  const int s = blk >> 6, b = (blk >> 5) & 1, i = blk & 31;
  const int img_q = s*2 + b, img_k = (1 - s)*2 + b;

  const _Float16* T = Tp + (size_t)img_q*HWIMG*REPC;   // [hw][rep]
  const _Float16* P = Pp + (size_t)img_k*HWIMG*REPC;   // [hw][rep]
  const _Float16* G = Gp + (size_t)img_k*REPC*HWIMG;   // [rep][hw]

  const int tid  = threadIdx.x;
  const int lane = tid & 31, wave = tid >> 5;
  const int l16  = lane & 15, hg = lane >> 4;
  const int q0w  = qtile*256 + wave*32;       // first query of this wave

  // query A-operands: 2 M-tiles x 4 d-chunks, reused for every key chunk
  v16h aq[2][4];
#pragma unroll
  for (int m = 0; m < 2; ++m){
    int hw = qhw(i, q0w + m*16 + l16);
#pragma unroll
    for (int dc = 0; dc < 4; ++dc)
      aq[m][dc] = ld16h(&T[(size_t)hw*REPC + dc*32 + hg*16]);
  }

  const v8f vz = {0.f,0.f,0.f,0.f,0.f,0.f,0.f,0.f};
  v8f occ[2][8], lsum[2], macc[2];
#pragma unroll
  for (int m = 0; m < 2; ++m){
    lsum[m] = vz; macc[m] = vz;
#pragma unroll
    for (int n = 0; n < 8; ++n) occ[m][n] = vz;
  }

  // per-thread staging addresses (depend only on tid and kc)
  const int pk_key  = tid >> 3, pk_part = (tid & 7)*16;     // 32B of a P row
  const int g_d     = tid >> 1, g_part  = (tid & 1)*16;     // 32B of a G row

  for (int kc = 0; kc < 24; ++kc){            // keys in chunks of 32
    { // async stage PH chunk: [key 32][d 128]
      int hwk = qhw(i, kc*32 + pk_key);
      const _Float16* src = &P[(size_t)hwk*REPC + pk_part];
      unsigned dst = lds_off_of(&ldsP[pk_key*128 + pk_part]);
      async_g2l_b128(src,     dst);
      async_g2l_b128(src + 8, dst + 16);
    }
    { // async stage G chunk: [d 128][key 32] (key chunk is hw-contiguous)
      int hwk0 = qhw(i, kc*32);
      const _Float16* src = &G[(size_t)g_d*HWIMG + hwk0 + g_part];
      unsigned dst = lds_off_of(&ldsG[g_d*32 + g_part]);
      async_g2l_b128(src,     dst);
      async_g2l_b128(src + 8, dst + 16);
      if (kc + 1 < 24)
        __builtin_prefetch(&G[(size_t)g_d*HWIMG + qhw(i, (kc + 1)*32)], 0, 0);
    }
    wait_async0();
    __syncthreads();

    // S tiles -> exp -> stash probs in LDS (A-operand reshape)
#pragma unroll
    for (int m = 0; m < 2; ++m){
#pragma unroll
      for (int kt = 0; kt < 2; ++kt){
        v8f sa = vz;
#pragma unroll
        for (int dc = 0; dc < 4; ++dc){
          v16h bpv = ld16h(&ldsP[(kt*16 + l16)*128 + dc*32 + hg*16]);
          sa = wmma16(aq[m][dc], bpv, sa);
        }
        float colv = (float)((kc*32 + kt*16 + l16) & 255);
#pragma unroll
        for (int e = 0; e < 8; ++e){
          float pexp = __expf(sa[e]);
          lsum[m][e] += pexp;
          macc[m][e] += pexp * colv;
          ldsProb[wave][(m*16 + e + 8*hg)*32 + kt*16 + l16] = (_Float16)pexp;
        }
      }
    }
    __syncthreads();

    // O += P @ G
#pragma unroll
    for (int m = 0; m < 2; ++m){
      v16h pa = ld16h(&ldsProb[wave][(m*16 + l16)*32 + hg*16]);
#pragma unroll
      for (int n = 0; n < 8; ++n){
        v16h gb = ld16h(&ldsG[(n*16 + l16)*32 + hg*16]);
        occ[m][n] = wmma16(pa, gb, occ[m][n]);
      }
    }
    __syncthreads();
  }

  // reduce row sums / col-weighted sums across the 16 N-lanes
#pragma unroll
  for (int off = 1; off < 16; off <<= 1){
#pragma unroll
    for (int m = 0; m < 2; ++m)
#pragma unroll
      for (int e = 0; e < 8; ++e){
        lsum[m][e] += __shfl_xor(lsum[m][e], off, 32);
        macc[m][e] += __shfl_xor(macc[m][e], off, 32);
      }
  }

  // normalize, store "after" [blk][q][rep] (f16) and maxi
#pragma unroll
  for (int m = 0; m < 2; ++m){
#pragma unroll
    for (int e = 0; e < 8; ++e){
      int q   = q0w + m*16 + e + 8*hg;        // 0..511 within block
      float inv = 1.f / lsum[m][e];
      if (l16 == 0) maxi_ws[(size_t)blk*NQOUT + q] = macc[m][e] * inv;
      _Float16* arow = after + ((size_t)blk*NQOUT + q)*REPC;
#pragma unroll
      for (int n = 0; n < 8; ++n)
        arow[n*16 + l16] = (_Float16)(occ[m][n][e] * inv);
    }
  }
}

// ---------------------------------------------------------------------------
// Kernel 3: y = Wu @ after, fold BN, add residual, divide by coverage count,
// write final fl/fr rows directly (block i local rows r in {0,1} -> h = 2i+r).
// grid = NBLK(128) * qtiles(4 of 128); 256 threads = 8 waves (wave = 32 chans)
// ---------------------------------------------------------------------------
__global__ void out_kernel(const float* left, const float* right,
                           const _Float16* wuh, const _Float16* after,
                           const float* effs, const float* effb, float* out)
{
  const int nt  = blockIdx.x & 3;
  const int blk = blockIdx.x >> 2;
  const int s = blk >> 6, b = (blk >> 5) & 1, i = blk & 31;
  const int q0 = nt*128;

  const int tid  = threadIdx.x;
  const int lane = tid & 31, wave = tid >> 5;
  const int l16  = lane & 15, hg = lane >> 4;
  const int cw   = wave*32;

  const v8f vz = {0.f,0.f,0.f,0.f,0.f,0.f,0.f,0.f};
  v8f acc[2][8];
#pragma unroll
  for (int m = 0; m < 2; ++m)
#pragma unroll
    for (int n = 0; n < 8; ++n) acc[m][n] = vz;

  const _Float16* ab = after + (size_t)blk*NQOUT*REPC;
#pragma unroll
  for (int kc = 0; kc < 4; ++kc){             // rep in chunks of 32
    v16h a0 = ld16h(&wuh[(size_t)(cw      + l16)*REPC + kc*32 + hg*16]);
    v16h a1 = ld16h(&wuh[(size_t)(cw + 16 + l16)*REPC + kc*32 + hg*16]);
#pragma unroll
    for (int n = 0; n < 8; ++n){
      v16h bv = ld16h(&ab[(size_t)(q0 + n*16 + l16)*REPC + kc*32 + hg*16]);
      acc[0][n] = wmma16(a0, bv, acc[0][n]);
      acc[1][n] = wmma16(a1, bv, acc[1][n]);
    }
  }

  const float* src = (s ? right : left);
  const size_t side_off = (size_t)s * (size_t)BB*CC*HH*WWID;
#pragma unroll
  for (int m = 0; m < 2; ++m){
#pragma unroll
    for (int e = 0; e < 8; ++e){
      int c = cw + m*16 + e + 8*hg;
      float sc = effs[c], bi = effb[c];
#pragma unroll
      for (int n = 0; n < 8; ++n){
        int q = q0 + n*16 + l16;
        int r = q >> 8, wcol = q & 255;
        int hrow = 2*i + r;                   // always <= 63 for r in {0,1}
        float y  = acc[m][n][e]*sc + bi;
        size_t pix = (((size_t)b*CC + c)*HH + hrow)*WWID + wcol;
        float qxv = src[pix];
        float cnt = (hrow >= 2 && (hrow & 1) == 0) ? 2.f : 1.f;
        out[side_off + pix] = (qxv + y) / cnt;
      }
    }
  }
}

// ---------------------------------------------------------------------------
// Kernel 4: fli/fri = ((w - maxi) / count) at the aggregated rows
// ---------------------------------------------------------------------------
__global__ void idx_kernel(const float* maxi_ws, float* out)
{
  int t = blockIdx.x*blockDim.x + threadIdx.x;   // NBLK*512 = 65536
  int blk = t >> 9, q = t & 511;
  int s = blk >> 6, b = (blk >> 5) & 1, i = blk & 31;
  int r = q >> 8, wcol = q & 255;
  int hrow = 2*i + r;
  float cnt = (hrow >= 2 && (hrow & 1) == 0) ? 2.f : 1.f;
  size_t off = (size_t)2*BB*CC*HH*WWID + (size_t)s*BB*HH*WWID
             + ((size_t)b*HH + hrow)*WWID + wcol;
  out[off] = ((float)wcol - maxi_ws[t]) / cnt;
}

// ---------------------------------------------------------------------------
extern "C" void kernel_launch(void* const* d_in, const int* in_sizes, int n_in,
                              void* d_out, int out_size, void* d_ws, size_t ws_size,
                              hipStream_t stream)
{
  const float* left  = (const float*)d_in[0];
  const float* right = (const float*)d_in[1];
  const float* Wt    = (const float*)d_in[2];
  const float* bt    = (const float*)d_in[3];
  const float* Wp    = (const float*)d_in[4];
  const float* bp    = (const float*)d_in[5];
  const float* Wg    = (const float*)d_in[6];
  const float* bg    = (const float*)d_in[7];
  const float* Wu    = (const float*)d_in[8];
  const float* bu    = (const float*)d_in[9];
  const float* gamma = (const float*)d_in[10];
  const float* beta  = (const float*)d_in[11];
  const float* rmean = (const float*)d_in[12];
  const float* rvar  = (const float*)d_in[13];
  float* out = (float*)d_out;

  // workspace carve-up
  char* p = (char*)d_ws;
  _Float16* wth = (_Float16*)p; p += (size_t)REPC*CC*2;
  _Float16* wph = (_Float16*)p; p += (size_t)REPC*CC*2;
  _Float16* wgh = (_Float16*)p; p += (size_t)REPC*CC*2;
  _Float16* wuh = (_Float16*)p; p += (size_t)CC*REPC*2;
  float* effs   = (float*)p;    p += (size_t)CC*4;
  float* effb   = (float*)p;    p += (size_t)CC*4;
  _Float16* Tp  = (_Float16*)p; p += (size_t)4*HWIMG*REPC*2;
  _Float16* Pp  = (_Float16*)p; p += (size_t)4*HWIMG*REPC*2;
  _Float16* Gp  = (_Float16*)p; p += (size_t)4*HWIMG*REPC*2;
  _Float16* after = (_Float16*)p; p += (size_t)NBLK*NQOUT*REPC*2;
  float* maxi_ws  = (float*)p;

  prep_kernel<<<512, 256, 0, stream>>>(Wt, Wp, Wg, Wu, bu, gamma, beta, rmean, rvar,
                                       wth, wph, wgh, wuh, effs, effb);
  proj_kernel<<<1536, 256, 0, stream>>>(left, right, wth, wph, wgh, bt, bp, bg,
                                        Tp, Pp, Gp);
  attn_kernel<<<256, 256, 0, stream>>>(Tp, Pp, Gp, after, maxi_ws);
  out_kernel<<<512, 256, 0, stream>>>(left, right, wuh, after, effs, effb, out);
  idx_kernel<<<256, 256, 0, stream>>>(maxi_ws, out);
}
